// SingleStepActionDecoder_1116691497582
// MI455X (gfx1250) — matrix operand order repaired and checked by reading, not compiled, not run on hardware
//
#include <hip/hip_runtime.h>
#include <hip/hip_bf16.h>

// ---------------------------------------------------------------------------
// Types for WMMA fragments (wave32, v_wmma_f32_16x16x32_f16)
// ---------------------------------------------------------------------------
typedef _Float16 h8   __attribute__((ext_vector_type(8)));
typedef _Float16 v16h __attribute__((ext_vector_type(16)));
typedef float    v8f  __attribute__((ext_vector_type(8)));
typedef float    f8   __attribute__((ext_vector_type(8)));

#define WMMA_F16(a, b, c) \
  __builtin_amdgcn_wmma_f32_16x16x32_f16(false, (a), false, (b), (short)0, (c), false, false)

// Problem constants
#define BATCH 32
#define NNODE 128
#define DIN   64
#define EDGES 16256          // 128*127
#define MHID  256
#define MOUT  64
#define NHID  256

// ---------------------------------------------------------------------------
// Fragment builders (contiguous K order, used for both A and B)
// A 16x32 f16: lane L holds row m=L&15; element j -> K=K0+(L>>4)*16+j
//              (16 consecutive K, matching b_frag).  Two contiguous 16B runs.
// ---------------------------------------------------------------------------
__device__ __forceinline__ v16h a_frag_f16(const _Float16* rowp, int K0, int half) {
  const h8 lo = *reinterpret_cast<const h8*>(rowp + K0 + half * 16);
  const h8 hi = *reinterpret_cast<const h8*>(rowp + K0 + half * 16 + 8);
  v16h a;
#pragma unroll
  for (int i = 0; i < 8; ++i) { a[i] = lo[i]; a[i + 8] = hi[i]; }
  return a;
}

__device__ __forceinline__ v16h a_frag_f32(const float* rowp, int K0, int half) {
  const f8 lo = *reinterpret_cast<const f8*>(rowp + K0 + half * 16);
  const f8 hi = *reinterpret_cast<const f8*>(rowp + K0 + half * 16 + 8);
  v16h a;
#pragma unroll
  for (int i = 0; i < 8; ++i) { a[i] = (_Float16)lo[i]; a[i + 8] = (_Float16)hi[i]; }
  return a;
}

// B 32x16 f16: lane L holds column n=L&15, 16 consecutive K starting at
// K0+(L>>4)*16.  With weights stored row-major [fan_out][fan_in] this is a
// single contiguous 32-byte chunk of row n.
__device__ __forceinline__ v16h b_frag(const _Float16* w, int ld, int n, int kbase) {
  return *reinterpret_cast<const v16h*>(w + (size_t)n * ld + kbase);
}

// ---------------------------------------------------------------------------
// Kernel 0: f32 -> f16 conversion (inputs + weights, once per launch)
// ---------------------------------------------------------------------------
__global__ void cvt_f32_f16(const float* __restrict__ src, _Float16* __restrict__ dst, int n) {
  int i = blockIdx.x * 256 + threadIdx.x;
  if (i < n) dst[i] = (_Float16)src[i];
}

// ---------------------------------------------------------------------------
// Kernel 1: fused edge MLP + per-receiver aggregation.
// Block = (b, recv r).  127 edges per block, 8 row-tiles of 16 (last row
// masked via scale=0).  Edges with r<64 use w1/w2, else w3/w4.
// agg[b*128+r][0..63] written once, fixed-order reduction (deterministic).
// ---------------------------------------------------------------------------
__global__ __launch_bounds__(128) void edge_kernel(
    const _Float16* __restrict__ xh,        // [B*128][64] f16 inputs
    const float*    __restrict__ rel_type,  // [B][E][2]
    const _Float16* __restrict__ w1h, const float* __restrict__ b1,
    const _Float16* __restrict__ w2h, const float* __restrict__ b2,
    const _Float16* __restrict__ w3h, const float* __restrict__ b3,
    const _Float16* __restrict__ w4h, const float* __restrict__ b4,
    float* __restrict__ agg)                // [B*128][64]
{
  __shared__ alignas(32) _Float16 hid[4][16 * MHID]; // per-wave 16x256 hidden
  __shared__ float part[8][64];                      // fixed-order reduction

  const int blk  = blockIdx.x;           // b*128 + r
  const int b    = blk >> 7;
  const int r    = blk & 127;
  const int wave = threadIdx.x >> 5;
  const int lane = threadIdx.x & 31;
  const int nl   = lane & 15;
  const int half = lane >> 4;

  const _Float16* wA; const float* bA; const _Float16* wB; const float* bB;
  if (r < 64) { wA = w1h; bA = b1; wB = w2h; bB = b2; }
  else        { wA = w3h; bA = b3; wB = w4h; bB = b4; }

  _Float16* H = hid[wave];
  float acc_agg[4] = {0.f, 0.f, 0.f, 0.f};

  for (int t = wave; t < 8; t += 4) {
    // ---- GEMM1: X(16x64) @ W^T -> hidden(16x256), relu, to LDS ----
    // A row for this lane: edge k = t*16 + nl of block; sender s skips r.
    int k = t * 16 + nl;
    int s = (k < r) ? k : k + 1;
    if (s > 127) s = 127;                          // masked row (scale=0 later)
    const _Float16* xrow = xh + (size_t)(b * NNODE + s) * DIN;
    v16h a0 = a_frag_f16(xrow, 0, half);
    v16h a1 = a_frag_f16(xrow, 32, half);

#pragma unroll
    for (int nt = 0; nt < 16; ++nt) {
      v8f acc = {0.f, 0.f, 0.f, 0.f, 0.f, 0.f, 0.f, 0.f};
      acc = WMMA_F16(a0, b_frag(wA, DIN, nt * 16 + nl, half * 16), acc);
      acc = WMMA_F16(a1, b_frag(wA, DIN, nt * 16 + nl, 32 + half * 16), acc);
      float bias = bA[nt * 16 + nl];
#pragma unroll
      for (int rr = 0; rr < 8; ++rr) {
        float h = acc[rr] + bias;
        h = h > 0.f ? h : 0.f;
        H[(rr + 8 * half) * MHID + nt * 16 + nl] = (_Float16)h;
      }
    }

    // per-row scales: rows of D-fragment are (rr + 8*half)
    float scale[8];
#pragma unroll
    for (int rr = 0; rr < 8; ++rr) {
      int kk = t * 16 + rr + 8 * half;
      scale[rr] = (kk < 127)
          ? rel_type[((size_t)b * EDGES + (size_t)r * 127 + kk) * 2 + 1]
          : 0.f;
    }

    // ---- GEMM2: hidden(16x256) @ W2^T -> msg(16x64), relu, scaled row-sum ----
    v16h a2[8];
#pragma unroll
    for (int ks = 0; ks < 8; ++ks)
      a2[ks] = a_frag_f16(H + nl * MHID, ks * 32, half);

#pragma unroll
    for (int nt = 0; nt < 4; ++nt) {
      v8f acc = {0.f, 0.f, 0.f, 0.f, 0.f, 0.f, 0.f, 0.f};
#pragma unroll
      for (int ks = 0; ks < 8; ++ks)
        acc = WMMA_F16(a2[ks], b_frag(wB, MHID, nt * 16 + nl, ks * 32 + half * 16), acc);
      float bias = bB[nt * 16 + nl];
      float p = 0.f;
#pragma unroll
      for (int rr = 0; rr < 8; ++rr) {
        float v = acc[rr] + bias;
        v = v > 0.f ? v : 0.f;
        p += v * scale[rr];
      }
      acc_agg[nt] += p;
    }
  }

#pragma unroll
  for (int nt = 0; nt < 4; ++nt)
    part[wave * 2 + half][nt * 16 + nl] = acc_agg[nt];
  __syncthreads();

  if (threadIdx.x < 64) {
    float s = 0.f;
#pragma unroll
    for (int i = 0; i < 8; ++i) s += part[i][threadIdx.x];
    agg[(size_t)blk * 64 + threadIdx.x] = s;
  }
}

// ---------------------------------------------------------------------------
// Kernel 2: node MLP 64->256->256->64 + residual; drop last node per batch.
// One wave per 16-row tile; 2 waves / block.
// ---------------------------------------------------------------------------
__global__ __launch_bounds__(64) void node_kernel(
    const float* __restrict__ agg,      // [4096][64]
    const float* __restrict__ inputs,   // [4096][64] f32
    const _Float16* __restrict__ wo1h, const float* __restrict__ bo1,
    const _Float16* __restrict__ wo2h, const float* __restrict__ bo2,
    const _Float16* __restrict__ wo3h, const float* __restrict__ bo3,
    float* __restrict__ out)            // [32*127*64]
{
  __shared__ alignas(32) _Float16 h1[2][16 * NHID];
  __shared__ alignas(32) _Float16 h2[2][16 * NHID];

  const int wave = threadIdx.x >> 5;
  const int lane = threadIdx.x & 31;
  const int nl   = lane & 15;
  const int half = lane >> 4;
  const int tile = blockIdx.x * 2 + wave;   // 0..255
  const int g    = tile * 16 + nl;          // this lane's A row

  _Float16* H1 = h1[wave];
  _Float16* H2 = h2[wave];

  // ---- stage A: agg(16x64) @ wo1^T -> h1(16x256) ----
  const float* arow = agg + (size_t)g * MOUT;
  v16h a0 = a_frag_f32(arow, 0, half);
  v16h a1 = a_frag_f32(arow, 32, half);
#pragma unroll
  for (int nt = 0; nt < 16; ++nt) {
    v8f acc = {0.f, 0.f, 0.f, 0.f, 0.f, 0.f, 0.f, 0.f};
    acc = WMMA_F16(a0, b_frag(wo1h, MOUT, nt * 16 + nl, half * 16), acc);
    acc = WMMA_F16(a1, b_frag(wo1h, MOUT, nt * 16 + nl, 32 + half * 16), acc);
    float bias = bo1[nt * 16 + nl];
#pragma unroll
    for (int rr = 0; rr < 8; ++rr) {
      float h = acc[rr] + bias;
      h = h > 0.f ? h : 0.f;
      H1[(rr + 8 * half) * NHID + nt * 16 + nl] = (_Float16)h;
    }
  }

  // ---- stage B: h1(16x256) @ wo2^T -> h2(16x256) ----
  v16h af[8];
#pragma unroll
  for (int ks = 0; ks < 8; ++ks)
    af[ks] = a_frag_f16(H1 + nl * NHID, ks * 32, half);
#pragma unroll
  for (int nt = 0; nt < 16; ++nt) {
    v8f acc = {0.f, 0.f, 0.f, 0.f, 0.f, 0.f, 0.f, 0.f};
#pragma unroll
    for (int ks = 0; ks < 8; ++ks)
      acc = WMMA_F16(af[ks], b_frag(wo2h, NHID, nt * 16 + nl, ks * 32 + half * 16), acc);
    float bias = bo2[nt * 16 + nl];
#pragma unroll
    for (int rr = 0; rr < 8; ++rr) {
      float h = acc[rr] + bias;
      h = h > 0.f ? h : 0.f;
      H2[(rr + 8 * half) * NHID + nt * 16 + nl] = (_Float16)h;
    }
  }

  // ---- stage C: h2(16x256) @ wo3^T -> pred(16x64) + residual ----
#pragma unroll
  for (int ks = 0; ks < 8; ++ks)
    af[ks] = a_frag_f16(H2 + nl * NHID, ks * 32, half);
#pragma unroll
  for (int nt = 0; nt < 4; ++nt) {
    v8f acc = {0.f, 0.f, 0.f, 0.f, 0.f, 0.f, 0.f, 0.f};
#pragma unroll
    for (int ks = 0; ks < 8; ++ks)
      acc = WMMA_F16(af[ks], b_frag(wo3h, NHID, nt * 16 + nl, ks * 32 + half * 16), acc);
    float bias = bo3[nt * 16 + nl];
#pragma unroll
    for (int rr = 0; rr < 8; ++rr) {
      int gg = tile * 16 + rr + 8 * half;     // global row of this D element
      int n  = gg & 127;
      int bb = gg >> 7;
      if (n != 127) {                          // out[:, :-1, :]
        int d = nt * 16 + nl;
        out[((size_t)(bb * 127 + n)) * MOUT + d] =
            acc[rr] + bias + inputs[(size_t)gg * DIN + d];
      }
    }
  }
}

// ---------------------------------------------------------------------------
// Host: workspace layout + launches (all on `stream`, graph-capture safe)
// ---------------------------------------------------------------------------
extern "C" void kernel_launch(void* const* d_in, const int* in_sizes, int n_in,
                              void* d_out, int out_size, void* d_ws, size_t ws_size,
                              hipStream_t stream) {
  const float* inputs   = (const float*)d_in[0];   // [32][128][64]
  const float* rel_type = (const float*)d_in[1];   // [32][16256][2]
  // d_in[2] rel_rec, d_in[3] rel_send, d_in[4] indices: structure hardcoded
  const float* w1  = (const float*)d_in[5];  const float* b1  = (const float*)d_in[6];
  const float* w2  = (const float*)d_in[7];  const float* b2  = (const float*)d_in[8];
  const float* w3  = (const float*)d_in[9];  const float* b3  = (const float*)d_in[10];
  const float* w4  = (const float*)d_in[11]; const float* b4  = (const float*)d_in[12];
  const float* wo1 = (const float*)d_in[13]; const float* bo1 = (const float*)d_in[14];
  const float* wo2 = (const float*)d_in[15]; const float* bo2 = (const float*)d_in[16];
  const float* wo3 = (const float*)d_in[17]; const float* bo3 = (const float*)d_in[18];

  char* ws = (char*)d_ws;
  size_t off = 0;
  auto carve = [&](size_t bytes) { char* p = ws + off; off += (bytes + 255) & ~(size_t)255; return p; };

  _Float16* xh   = (_Float16*)carve((size_t)BATCH * NNODE * DIN * 2);
  _Float16* w1h  = (_Float16*)carve((size_t)MHID * DIN * 2);
  _Float16* w2h  = (_Float16*)carve((size_t)MOUT * MHID * 2);
  _Float16* w3h  = (_Float16*)carve((size_t)MHID * DIN * 2);
  _Float16* w4h  = (_Float16*)carve((size_t)MOUT * MHID * 2);
  _Float16* wo1h = (_Float16*)carve((size_t)NHID * MOUT * 2);
  _Float16* wo2h = (_Float16*)carve((size_t)NHID * NHID * 2);
  _Float16* wo3h = (_Float16*)carve((size_t)DIN * NHID * 2);
  float*    agg  = (float*)carve((size_t)BATCH * NNODE * MOUT * 4);
  (void)ws_size; (void)in_sizes; (void)n_in; (void)out_size;

  auto cvt = [&](const float* s, _Float16* d, int n) {
    cvt_f32_f16<<<(n + 255) / 256, 256, 0, stream>>>(s, d, n);
  };
  cvt(inputs, xh, BATCH * NNODE * DIN);
  cvt(w1, w1h, MHID * DIN);
  cvt(w2, w2h, MOUT * MHID);
  cvt(w3, w3h, MHID * DIN);
  cvt(w4, w4h, MOUT * MHID);
  cvt(wo1, wo1h, NHID * MOUT);
  cvt(wo2, wo2h, NHID * NHID);
  cvt(wo3, wo3h, DIN * NHID);

  edge_kernel<<<BATCH * NNODE, 128, 0, stream>>>(
      xh, rel_type, w1h, b1, w2h, b2, w3h, b3, w4h, b4, agg);

  node_kernel<<<(BATCH * NNODE) / 16 / 2, 64, 0, stream>>>(
      agg, inputs, wo1h, bo1, wo2h, bo2, wo3h, bo3, (float*)d_out);
}
